// SAGE_my_78365973283346
// MI455X (gfx1250) — matrix operand order repaired and checked by reading, not compiled, not run on hardware
//
#include <hip/hip_runtime.h>

typedef __attribute__((ext_vector_type(16))) _Float16 v16h;
typedef __attribute__((ext_vector_type(8)))  float    v8f;

#define D 256            // all feature dims are 256
#define K2 512           // fused K = 2*D  ([h | mean] @ [Ws;Wn])
#define BM 128
#define BN 64
#define BK 32
#define NSTEP (K2 / BK)  // 16 k-steps
#define LDA (BK + 8)     // 40 halves = 80B row stride: 16B aligned, conflict-free
#define LDB (K2 + 16)    // 528 halves = 1056B row stride: 16B aligned

// ---------------- utility kernels ----------------

__global__ __launch_bounds__(256) void k_zero(float* __restrict__ p, int n) {
    int i = blockIdx.x * blockDim.x + threadIdx.x;
    int stride = gridDim.x * blockDim.x;
    for (; i < n; i += stride) p[i] = 0.0f;
}

__global__ __launch_bounds__(256) void k_deg(const int* __restrict__ dst,
                                             float* __restrict__ deg, int E) {
    int e = blockIdx.x * blockDim.x + threadIdx.x;
    if (e < E) atomicAdd(&deg[dst[e]], 1.0f);
}

__global__ __launch_bounds__(256) void k_invdeg(const float* __restrict__ deg,
                                                float* __restrict__ invdeg, int n) {
    int i = blockIdx.x * blockDim.x + threadIdx.x;
    if (i < n) invdeg[i] = 1.0f / fmaxf(deg[i], 1.0f);
}

// Build column-major f16 weight panel: W16T[col][k], k<256 from Ws, k>=256 from Wn.
__global__ __launch_bounds__(256) void k_convw(const float* __restrict__ Ws,
                                               const float* __restrict__ Wn,
                                               _Float16* __restrict__ W16T) {
    int idx = blockIdx.x * blockDim.x + threadIdx.x;   // 0 .. 512*256-1
    if (idx < K2 * D) {
        int col = idx & (D - 1);
        int k   = idx >> 8;
        float v = (k < D) ? Ws[k * D + col] : Wn[(k - D) * D + col];
        W16T[(long long)col * K2 + k] = (_Float16)v;
    }
}

// one wave per edge: lanes cover the 256 columns (8 each), contiguous atomics
__global__ __launch_bounds__(256) void k_scatter(const float* __restrict__ h,
                                                 const int* __restrict__ src,
                                                 const int* __restrict__ dst,
                                                 float* __restrict__ agg, int E) {
    long long idx = (long long)blockIdx.x * blockDim.x + threadIdx.x;
    int e = (int)(idx >> 5);
    if (e >= E) return;
    int c = (int)(idx & 31) * 8;
    const float* hp = h + (long long)src[e] * D + c;
    float* ap = agg + (long long)dst[e] * D + c;
#pragma unroll
    for (int j = 0; j < 8; ++j) atomicAdd(ap + j, hp[j]);
}

// ---------------- WMMA GEMM: OUT = [X | agg*invdeg] @ W + bias ----------------

__global__ __launch_bounds__(256) void k_gemm_wmma(
    const float* __restrict__ X,        // [nrows, 256]
    const float* __restrict__ AGG,      // [nrows, 256]
    const float* __restrict__ invdeg,   // [nrows]
    const _Float16* __restrict__ W16T,  // [256 cols][512 k] column-major
    const float* __restrict__ bias,     // [256]
    float* __restrict__ OUT,            // [nrows, 256]
    int nrows) {
    __shared__ _Float16 As[2][BM][LDA]; // double-buffered A tile
    __shared__ _Float16 Bp[BN][LDB];    // whole 64-col x 512-k weight panel

    const int tid  = threadIdx.x;
    const int lane = tid & 31;
    const int wave = tid >> 5;
    const int wr   = wave & 3;          // 4 row-groups of 32
    const int wc   = wave >> 2;         // 2 col-groups of 32
    const int hf   = lane >> 4;         // lane half (0: lanes 0-15, 1: 16-31)
    const int l15  = lane & 15;
    const int row0 = blockIdx.x * BM;
    const int col0 = blockIdx.y * BN;

    // --- preload entire B panel: 64 cols x 512 k (contiguous 16B on both sides)
#pragma unroll
    for (int chunk = 0; chunk < 16; ++chunk) {
        int idx8 = chunk * 256 + tid;          // 8-half groups
        int col  = idx8 >> 6;                  // 0..63
        int kc   = (idx8 & 63) * 8;            // 0..504
        const _Float16* p = W16T + (long long)(col0 + col) * K2 + kc;
        _Float16* q = &Bp[col][kc];
#pragma unroll
        for (int j = 0; j < 8; ++j) q[j] = p[j];
    }

    // --- hoist per-thread staging state (rows are fixed for whole kernel)
    const int cstage = (tid & 3) * 8;          // staged k-offset within tile
    int       srow[2];
    long long rowoff[2];
    float     scX[2], scA[2];
#pragma unroll
    for (int it = 0; it < 2; ++it) {
        int grow = row0 + (tid >> 2) + it * 64;
        bool ok  = grow < nrows;
        int crow = ok ? grow : (nrows - 1);    // clamp: reads stay in-bounds
        srow[it]   = (tid >> 2) + it * 64;
        rowoff[it] = (long long)crow * D;
        scX[it]    = ok ? 1.0f : 0.0f;
        scA[it]    = ok ? invdeg[crow] : 0.0f;
    }

    // branch-free A tile staging: source/scale via selects only
    auto stage = [&](int s, int bufi) {
        int k0 = s * BK;
#pragma unroll
        for (int it = 0; it < 2; ++it) {
            int gk = k0 + cstage;              // 0..511, never straddles 256
            int kk = gk & (D - 1);
            bool ph = gk >= D;
            const float* p = (ph ? AGG : X) + rowoff[it] + kk;
            float sc = ph ? scA[it] : scX[it];
            float v[8];
#pragma unroll
            for (int j = 0; j < 8; ++j) v[j] = p[j] * sc;
#pragma unroll
            for (int j = 0; j < 8; ++j) As[bufi][srow[it]][cstage + j] = (_Float16)v[j];
        }
    };

    v8f acc[2][2];
    v8f zero = {};
#pragma unroll
    for (int a = 0; a < 2; ++a)
#pragma unroll
        for (int b = 0; b < 2; ++b) acc[a][b] = zero;

    stage(0, 0);
    __syncthreads();                           // covers panel preload + first tile

#pragma unroll 2
    for (int s = 0; s < NSTEP; ++s) {
        int bufi = s & 1;
        if (s + 1 < NSTEP) stage(s + 1, bufi ^ 1);   // prefetch next tile

        // --- WMMA: 2x2 16x16 tiles per wave, one 16x16x32 op each
        int k0 = s * BK;
        v16h afrag[2], bfrag[2];
#pragma unroll
        for (int mt = 0; mt < 2; ++mt) {
            const _Float16* ap = &As[bufi][wr * 32 + mt * 16 + l15][0];
#pragma unroll
            for (int j = 0; j < 8; ++j) {      // A layout: 16 contiguous K halves (k = 16*hf + s), matches B
                afrag[mt][j]     = ap[hf * 16 + j];
                afrag[mt][8 + j] = ap[hf * 16 + 8 + j];
            }
        }
#pragma unroll
        for (int nt = 0; nt < 2; ++nt) {
            const _Float16* bp = &Bp[wc * 32 + nt * 16 + l15][k0 + hf * 16];
#pragma unroll
            for (int j = 0; j < 16; ++j)       // B layout: 16 contiguous K halves (k = 16*hf + j), matches A
                bfrag[nt][j] = bp[j];
        }
#pragma unroll
        for (int mt = 0; mt < 2; ++mt)
#pragma unroll
            for (int nt = 0; nt < 2; ++nt)
                acc[mt][nt] = __builtin_amdgcn_wmma_f32_16x16x32_f16(
                    false, afrag[mt], false, bfrag[nt],
                    (short)0, acc[mt][nt], false, false);
        __syncthreads();                       // next-tile stores done, frags consumed
    }

    // --- epilogue: +bias, store (C/D layout: M = v + 8*hf, N = lane&15)
#pragma unroll
    for (int nt = 0; nt < 2; ++nt) {
        int n = col0 + wc * 32 + nt * 16 + l15;
        float bv = bias[n];
#pragma unroll
        for (int mt = 0; mt < 2; ++mt) {
#pragma unroll
            for (int v = 0; v < 8; ++v) {
                int m = row0 + wr * 32 + mt * 16 + hf * 8 + v;
                if (m < nrows) OUT[(long long)m * D + n] = acc[mt][nt][v] + bv;
            }
        }
    }
}

// ---------------- BatchNorm ----------------

__global__ __launch_bounds__(256) void k_bn_reduce(const float* __restrict__ h,
                                                   float* __restrict__ sums, // [512]
                                                   int nrows) {
    int col = threadIdx.x;                 // 256 cols
    int r0 = blockIdx.x * 128;
    int r1 = min(r0 + 128, nrows);
    float s = 0.0f, s2 = 0.0f;
    for (int r = r0; r < r1; ++r) {
        float v = h[(long long)r * D + col];
        s += v; s2 += v * v;
    }
    atomicAdd(&sums[col], s);
    atomicAdd(&sums[D + col], s2);
}

__global__ __launch_bounds__(256) void k_bn_apply(const float* __restrict__ lin,
                                                  const float* __restrict__ sums,
                                                  const float* __restrict__ g,
                                                  const float* __restrict__ be,
                                                  float* __restrict__ out,
                                                  int nrows, int act) {
    int row = blockIdx.x;
    int col = threadIdx.x;
    if (row >= nrows) return;
    float inv_n = 1.0f / (float)nrows;
    float mu  = sums[col] * inv_n;
    float var = sums[D + col] * inv_n - mu * mu;
    float rs  = rsqrtf(var + 1e-5f);
    float v = (lin[(long long)row * D + col] - mu) * rs * g[col] + be[col];
    v = (act == 0) ? fmaxf(v, 0.0f) : 1.0f / (1.0f + __expf(-v));
    out[(long long)row * D + col] = v;
}

// ---------------- host launch ----------------

extern "C" void kernel_launch(void* const* d_in, const int* in_sizes, int n_in,
                              void* d_out, int out_size, void* d_ws, size_t ws_size,
                              hipStream_t stream) {
    const float* x   = (const float*)d_in[0];
    const int*   src = (const int*)d_in[1];
    const int*   dst = (const int*)d_in[2];
    const int N = in_sizes[0] / D;
    const int E = in_sizes[1];

    // scratch layout
    char* base = (char*)d_ws;
    size_t off = 0;
    auto carve = [&](size_t bytes) { char* p = base + off; off += (bytes + 255) & ~(size_t)255; return p; };
    float*    deg    = (float*)carve((size_t)N * 4);
    float*    invdeg = (float*)carve((size_t)N * 4);
    float*    agg    = (float*)carve((size_t)N * D * 4);
    float*    lin    = (float*)carve((size_t)N * D * 4);
    float*    hbuf   = (float*)carve((size_t)N * D * 4);
    _Float16* W16T   = (_Float16*)carve((size_t)3 * K2 * D * 2);
    float*    sums   = (float*)carve((size_t)2 * D * 4);
    (void)ws_size;

    // degrees (shared across layers)
    k_zero<<<(N + 255) / 256, 256, 0, stream>>>(deg, N);
    k_deg<<<(E + 255) / 256, 256, 0, stream>>>(dst, deg, E);
    k_invdeg<<<(N + 255) / 256, 256, 0, stream>>>(deg, invdeg, N);

    // pre-convert weights to f16 column-major [Ws;Wn] per layer
    for (int i = 0; i < 3; ++i) {
        const float* Ws = (const float*)d_in[3 + 5 * i + 0];
        const float* Wn = (const float*)d_in[3 + 5 * i + 1];
        k_convw<<<(K2 * D + 255) / 256, 256, 0, stream>>>(Ws, Wn, W16T + (size_t)i * K2 * D);
    }

    const float* hcur = x;
    const int gemm_gx = (N + BM - 1) / BM;
    for (int i = 0; i < 3; ++i) {
        const float* b  = (const float*)d_in[3 + 5 * i + 2];
        const float* g  = (const float*)d_in[3 + 5 * i + 3];
        const float* be = (const float*)d_in[3 + 5 * i + 4];

        k_zero<<<4096, 256, 0, stream>>>(agg, N * D);
        long long sthreads = (long long)E * 32;
        k_scatter<<<(unsigned)((sthreads + 255) / 256), 256, 0, stream>>>(hcur, src, dst, agg, E);

        k_gemm_wmma<<<dim3(gemm_gx, D / BN), 256, 0, stream>>>(
            hcur, agg, invdeg, W16T + (size_t)i * K2 * D, b, lin, N);

        k_zero<<<2, 256, 0, stream>>>(sums, 2 * D);
        k_bn_reduce<<<(N + 127) / 128, 256, 0, stream>>>(lin, sums, N);

        float* hout = (i == 2) ? (float*)d_out : hbuf;
        k_bn_apply<<<N, 256, 0, stream>>>(lin, sums, g, be, hout, N, (i == 2) ? 1 : 0);
        hcur = hout;
    }
}
